// MultiHeadSelfAttention_4784593568459
// MI455X (gfx1250) — matrix operand order repaired and checked by reading, not compiled, not run on hardware
//
#include <hip/hip_runtime.h>

// ---------------------------------------------------------------------------
// MHA forward for MI455X (gfx1250): f16 WMMA (v_wmma_f32_16x16x32_f16) with
// f32 accumulation, flash-style online softmax, wave32 tiling.
// Round 3: 32x64 GEMM wave tiles (8 WMMA / 12 loads per k-step), K-fragment
// cross-iteration prefetch in attention, batched shuffle reductions.
// ---------------------------------------------------------------------------

typedef __attribute__((ext_vector_type(16))) _Float16 v16h;
typedef __attribute__((ext_vector_type(8)))  _Float16 v8h;
typedef __attribute__((ext_vector_type(8)))  float    v8f;

#define D_MODEL 1024
#define NHEAD   16
#define DK      64
#define SEQ     2048
#define BATCH   2
#define ROWS    (BATCH * SEQ)      // 4096 token rows

__device__ __forceinline__ v8f wmma_f16(v16h a, v16h b, v8f c) {
  return __builtin_amdgcn_wmma_f32_16x16x32_f16(
      /*neg_a=*/false, a, /*neg_b=*/false, b,
      /*c_mod=*/(short)0, c, /*reuse_a=*/false, /*reuse_b=*/false);
}

// A-fragment (16x32, f16): lane m = lane&15 is row m of the tile.
// K elements contiguous: offsets {hi*16 .. hi*16+15}.
__device__ __forceinline__ v16h ld_frag_a(const _Float16* __restrict__ base,
                                          int ld, int lane) {
  int m  = lane & 15;
  int hi = lane >> 4;
  const _Float16* p = base + (size_t)m * ld + hi * 16;
  v8h lo = *(const v8h*)(p);
  v8h hq = *(const v8h*)(p + 8);
  return __builtin_shufflevector(lo, hq, 0,1,2,3,4,5,6,7,8,9,10,11,12,13,14,15);
}

// B-fragment (32x16, f16) fed from row-major B^T: lane n = lane&15 holds
// column n of B (= row n of B^T); K contiguous at offset hi*16.
__device__ __forceinline__ v16h ld_frag_b(const _Float16* __restrict__ baseT,
                                          int ld, int lane) {
  int n  = lane & 15;
  int hi = lane >> 4;
  const _Float16* p = baseT + (size_t)n * ld + hi * 16;
  v8h lo = *(const v8h*)(p);
  v8h hq = *(const v8h*)(p + 8);
  return __builtin_shufflevector(lo, hq, 0,1,2,3,4,5,6,7,8,9,10,11,12,13,14,15);
}

// ---------------------------------------------------------------------------
// f32 -> f16 elementwise convert
// ---------------------------------------------------------------------------
__global__ __launch_bounds__(256) void cvt_f32_to_f16(
    const float* __restrict__ in, _Float16* __restrict__ out, int n) {
  int i = blockIdx.x * 256 + threadIdx.x;
  if (i < n) out[i] = (_Float16)in[i];
}

// ---------------------------------------------------------------------------
// W [dim x dim] f32 (in,out) -> W^T [dim x dim] f16 (out,in), tiled via LDS
// ---------------------------------------------------------------------------
__global__ __launch_bounds__(256) void transpose_to_f16(
    const float* __restrict__ W, _Float16* __restrict__ Wt, int dim) {
  __shared__ float tile[32][33];
  int x = blockIdx.x * 32 + threadIdx.x;        // column of W
  int y0 = blockIdx.y * 32;
  #pragma unroll
  for (int j = threadIdx.y; j < 32; j += 8)
    tile[j][threadIdx.x] = W[(size_t)(y0 + j) * dim + x];
  __syncthreads();
  int ox = blockIdx.y * 32 + threadIdx.x;       // column of W^T (= row of W)
  int oy0 = blockIdx.x * 32;
  #pragma unroll
  for (int j = threadIdx.y; j < 32; j += 8)
    Wt[(size_t)(oy0 + j) * dim + ox] = (_Float16)tile[threadIdx.x][j];
}

// ---------------------------------------------------------------------------
// GEMM: Y[M x N] = A[M x K] * Bt^T + bias.  A, Bt row-major f16 (Bt = B^T).
// Each wave owns a 32x64 output tile (2x4 WMMA accumulators): 8 WMMAs per
// 12 b128 loads per k-step, register double buffered.
// MODE 0: store f16 as [B,H,S,dk]      (Q, K)
// MODE 1: store f16 as [B,H,dk,S]      (V transposed)
// MODE 2: store f32 row-major [M x N]  (final output)
// ---------------------------------------------------------------------------
template <int MODE>
__global__ __launch_bounds__(128) void gemm_wmma(
    const _Float16* __restrict__ A, const _Float16* __restrict__ Bt,
    const float* __restrict__ bias, void* __restrict__ out,
    int Mrows, int Kdim, int Ncols) {
  int lane = threadIdx.x & 31;
  int wid  = threadIdx.x >> 5;
  int wtile = blockIdx.x * 4 + wid;
  int N64 = Ncols >> 6;
  int tm = wtile / N64;
  int tn = wtile % N64;
  int r0 = tm * 32, c0 = tn * 64;
  if (r0 >= Mrows) return;

  const _Float16* Ar[2];
  const _Float16* Bc[4];
  #pragma unroll
  for (int i = 0; i < 2; ++i) Ar[i] = A + (size_t)(r0 + i * 16) * Kdim;
  #pragma unroll
  for (int j = 0; j < 4; ++j) Bc[j] = Bt + (size_t)(c0 + j * 16) * Kdim;

  v8f acc[2][4] = {};
  v16h a[2], b[4];
  #pragma unroll
  for (int i = 0; i < 2; ++i) a[i] = ld_frag_a(Ar[i], Kdim, lane);
  #pragma unroll
  for (int j = 0; j < 4; ++j) b[j] = ld_frag_b(Bc[j], Kdim, lane);

  for (int k0 = 0; k0 + 32 < Kdim; k0 += 32) {
    // prefetch next k-step while WMMAs run on current fragments
    v16h na[2], nb[4];
    #pragma unroll
    for (int i = 0; i < 2; ++i) na[i] = ld_frag_a(Ar[i] + k0 + 32, Kdim, lane);
    #pragma unroll
    for (int j = 0; j < 4; ++j) nb[j] = ld_frag_b(Bc[j] + k0 + 32, Kdim, lane);
    #pragma unroll
    for (int i = 0; i < 2; ++i)
      #pragma unroll
      for (int j = 0; j < 4; ++j)
        acc[i][j] = wmma_f16(a[i], b[j], acc[i][j]);
    #pragma unroll
    for (int i = 0; i < 2; ++i) a[i] = na[i];
    #pragma unroll
    for (int j = 0; j < 4; ++j) b[j] = nb[j];
  }
  #pragma unroll
  for (int i = 0; i < 2; ++i)
    #pragma unroll
    for (int j = 0; j < 4; ++j)
      acc[i][j] = wmma_f16(a[i], b[j], acc[i][j]);

  int n = lane & 15, hi = lane >> 4;
  #pragma unroll
  for (int mt = 0; mt < 2; ++mt) {
    #pragma unroll
    for (int nt = 0; nt < 4; ++nt) {
      int ocol = c0 + nt * 16 + n;
      float bv = bias[ocol];
      #pragma unroll
      for (int r = 0; r < 8; ++r) {
        int row = r0 + mt * 16 + r + hi * 8;     // global token row (b*S + s)
        float val = acc[mt][nt][r] + bv;
        if (MODE == 2) {
          ((float*)out)[(size_t)row * Ncols + ocol] = val;
        } else {
          int bb = row >> 11;          // /SEQ
          int s = row & (SEQ - 1);
          int h = ocol >> 6;           // /DK
          int d = ocol & (DK - 1);
          _Float16* o16 = (_Float16*)out;
          if (MODE == 0)               // [B,H,S,dk]
            o16[(((size_t)(bb * NHEAD + h)) * SEQ + s) * DK + d] = (_Float16)val;
          else                          // [B,H,dk,S]
            o16[(((size_t)(bb * NHEAD + h)) * DK + d) * SEQ + s] = (_Float16)val;
        }
      }
    }
  }
}

// ---------------------------------------------------------------------------
// Flash attention: one wave per (b,h, 16-query block). Streams keys in blocks
// of 32, online softmax with lane-local partial denominators (cross-lane sum
// only at the end), P tile C-layout -> A-layout via per-wave LDS tile.
// K fragments for the next block are prefetched before the softmax work.
// ---------------------------------------------------------------------------
__global__ __launch_bounds__(128) void attn_flash(
    const _Float16* __restrict__ Qh,  // [B,H,S,dk]
    const _Float16* __restrict__ Kh,  // [B,H,S,dk]
    const _Float16* __restrict__ Vt,  // [B,H,dk,S]
    _Float16* __restrict__ Ah) {      // [B,S,H,dk]
  __shared__ _Float16 plds[4][16 * 32];
  int lane = threadIdx.x & 31;
  int wid  = threadIdx.x >> 5;
  int wave = blockIdx.x * 4 + wid;           // 4096 waves total
  int qb = (wave & 127) * 16;                // query block start
  int bh = wave >> 7;                        // b*NHEAD + h
  const float scale = 0.125f;                // 1/sqrt(dk)

  const _Float16* Qb = Qh + (size_t)bh * SEQ * DK + (size_t)qb * DK;
  const _Float16* Kb = Kh + (size_t)bh * SEQ * DK;
  const _Float16* Vb = Vt + (size_t)bh * DK * SEQ;

  // Q fragments (reused for every key block): dk=64 -> two K=32 chunks
  v16h aq0 = ld_frag_a(Qb, DK, lane);
  v16h aq1 = ld_frag_a(Qb + 32, DK, lane);

  float mrun[8], lsum[8];                    // lsum is LANE-LOCAL partial
  v8f O[4] = {};
  #pragma unroll
  for (int r = 0; r < 8; ++r) { mrun[r] = -1e30f; lsum[r] = 0.0f; }

  int n = lane & 15, hi = lane >> 4;
  _Float16* pl = plds[wid];

  // K fragments for the first key block
  v16h bk[4];
  bk[0] = ld_frag_b(Kb, DK, lane);
  bk[1] = ld_frag_b(Kb + 32, DK, lane);
  bk[2] = ld_frag_b(Kb + 16 * DK, DK, lane);
  bk[3] = ld_frag_b(Kb + 16 * DK + 32, DK, lane);

  for (int kb = 0; kb < SEQ; kb += 32) {
    // ---- scores for 32 keys: two 16x16 C tiles, each K=64 (2 chained WMMAs)
    v8f s0 = {}, s1 = {};
    s0 = wmma_f16(aq0, bk[0], s0);
    s0 = wmma_f16(aq1, bk[1], s0);
    s1 = wmma_f16(aq0, bk[2], s1);
    s1 = wmma_f16(aq1, bk[3], s1);

    // ---- issue V loads and NEXT-block K loads now; they overlap softmax
    v16h vf[4];
    #pragma unroll
    for (int t = 0; t < 4; ++t) {
      const _Float16* vp = Vb + (size_t)(t * 16 + n) * SEQ + kb + hi * 16;
      v8h lo = *(const v8h*)(vp);
      v8h hq = *(const v8h*)(vp + 8);
      vf[t] = __builtin_shufflevector(lo, hq,
                                      0,1,2,3,4,5,6,7,8,9,10,11,12,13,14,15);
    }
    {
      int kn = (kb + 32 < SEQ) ? kb + 32 : kb;   // redundant reload last iter
      const _Float16* Kn = Kb + (size_t)kn * DK;
      bk[0] = ld_frag_b(Kn, DK, lane);
      bk[1] = ld_frag_b(Kn + 32, DK, lane);
      bk[2] = ld_frag_b(Kn + 16 * DK, DK, lane);
      bk[3] = ld_frag_b(Kn + 16 * DK + 32, DK, lane);
    }

    // ---- row max: batched by stage so 8 ds_bpermutes are in flight at once
    float rm[8];
    #pragma unroll
    for (int r = 0; r < 8; ++r) rm[r] = fmaxf(s0[r], s1[r]);
    #pragma unroll
    for (int mask = 1; mask <= 8; mask <<= 1) {
      #pragma unroll
      for (int r = 0; r < 8; ++r)
        rm[r] = fmaxf(rm[r], __shfl_xor(rm[r], mask, 32));
    }

    // ---- online softmax update (fully unrolled; no cross-lane sum here)
    #pragma unroll
    for (int r = 0; r < 8; ++r) {
      float mn = fmaxf(mrun[r], rm[r]);
      float corr = __expf(scale * (mrun[r] - mn));
      float p0 = __expf(scale * (s0[r] - mn));
      float p1 = __expf(scale * (s1[r] - mn));
      lsum[r] = lsum[r] * corr + (p0 + p1);
      mrun[r] = mn;
      O[0][r] *= corr; O[1][r] *= corr; O[2][r] *= corr; O[3][r] *= corr;
      int row = r + hi * 8;                    // C-layout row
      pl[row * 32 + n]      = (_Float16)p0;    // keys kb .. kb+15
      pl[row * 32 + n + 16] = (_Float16)p1;    // keys kb+16 .. kb+31
    }

    // ---- P tile: reload from LDS in A-fragment layout (16x32, K = key dim)
    v16h pa;
    {
      int m = lane & 15;
      const _Float16* pp = &pl[m * 32 + hi * 16];
      v8h lo = *(const v8h*)(pp);
      v8h hq = *(const v8h*)(pp + 8);
      pa = __builtin_shufflevector(lo, hq,
                                   0,1,2,3,4,5,6,7,8,9,10,11,12,13,14,15);
    }

    // ---- O += P @ V
    #pragma unroll
    for (int t = 0; t < 4; ++t)
      O[t] = wmma_f16(pa, vf[t], O[t]);
  }

  // ---- final cross-lane reduction of the softmax denominator (once)
  #pragma unroll
  for (int mask = 1; mask <= 8; mask <<= 1) {
    #pragma unroll
    for (int r = 0; r < 8; ++r)
      lsum[r] += __shfl_xor(lsum[r], mask, 32);
  }

  // ---- epilogue: normalize and store to [B, S, H, dk] f16
  int b = bh >> 4, h = bh & 15;
  #pragma unroll
  for (int r = 0; r < 8; ++r) {
    float inv = 1.0f / lsum[r];
    int s = qb + r + hi * 8;
    size_t base = (((size_t)(b * SEQ + s)) * NHEAD + h) * DK;
    #pragma unroll
    for (int t = 0; t < 4; ++t)
      Ah[base + t * 16 + n] = (_Float16)(O[t][r] * inv);
  }
}

// ---------------------------------------------------------------------------
// Host-side orchestration
// ---------------------------------------------------------------------------
extern "C" void kernel_launch(void* const* d_in, const int* in_sizes, int n_in,
                              void* d_out, int out_size, void* d_ws, size_t ws_size,
                              hipStream_t stream) {
  const float* x  = (const float*)d_in[0];
  const float* Wq = (const float*)d_in[1];
  const float* bq = (const float*)d_in[2];
  const float* Wk = (const float*)d_in[3];
  const float* bk = (const float*)d_in[4];
  const float* Wv = (const float*)d_in[5];
  const float* bv = (const float*)d_in[6];
  const float* Wo = (const float*)d_in[7];
  const float* bo = (const float*)d_in[8];
  float* out = (float*)d_out;

  const size_t NX = (size_t)ROWS * D_MODEL;    // 4M elements
  const size_t NW = (size_t)D_MODEL * D_MODEL; // 1M elements
  _Float16* ws  = (_Float16*)d_ws;
  _Float16* xh  = ws;                // 4M
  _Float16* WqT = xh  + NX;          // 1M
  _Float16* WkT = WqT + NW;
  _Float16* WvT = WkT + NW;
  _Float16* WoT = WvT + NW;
  _Float16* Qh  = WoT + NW;          // 4M  [B,H,S,dk]
  _Float16* Kh  = Qh  + NX;          // 4M  [B,H,S,dk]
  _Float16* Vt  = Kh  + NX;          // 4M  [B,H,dk,S]
  _Float16* Ah  = Vt  + NX;          // 4M  [B,S,H,dk]

  // 1) convert activations to f16
  cvt_f32_to_f16<<<(int)(NX / 256), 256, 0, stream>>>(x, xh, (int)NX);

  // 2) transpose weights to f16 [out][in]
  dim3 tb(32, 8), tg(D_MODEL / 32, D_MODEL / 32);
  transpose_to_f16<<<tg, tb, 0, stream>>>(Wq, WqT, D_MODEL);
  transpose_to_f16<<<tg, tb, 0, stream>>>(Wk, WkT, D_MODEL);
  transpose_to_f16<<<tg, tb, 0, stream>>>(Wv, WvT, D_MODEL);
  transpose_to_f16<<<tg, tb, 0, stream>>>(Wo, WoT, D_MODEL);

  // 3) QKV projections: 32x64 wave tiles -> (4096/32)*(1024/64) = 2048 waves
  int gblocks = (ROWS / 32) * (D_MODEL / 64) / 4;  // 512
  gemm_wmma<0><<<gblocks, 128, 0, stream>>>(xh, WqT, bq, Qh, ROWS, D_MODEL, D_MODEL);
  gemm_wmma<0><<<gblocks, 128, 0, stream>>>(xh, WkT, bk, Kh, ROWS, D_MODEL, D_MODEL);
  gemm_wmma<1><<<gblocks, 128, 0, stream>>>(xh, WvT, bv, Vt, ROWS, D_MODEL, D_MODEL);

  // 4) flash attention: 4096 waves (B*H*128 query blocks), 4 waves/block
  attn_flash<<<1024, 128, 0, stream>>>(Qh, Kh, Vt, Ah);

  // 5) output projection -> f32 d_out
  gemm_wmma<2><<<gblocks, 128, 0, stream>>>(Ah, WoT, bo, out, ROWS, D_MODEL, D_MODEL);
}
